// MSAttBlock_2482491097151
// MI455X (gfx1250) — matrix operand order repaired and checked by reading, not compiled, not run on hardware
//
#include <hip/hip_runtime.h>

typedef __attribute__((ext_vector_type(16))) __bf16 v16bf;
typedef __attribute__((ext_vector_type(8)))  float  v8f;

#define H_  192
#define W_  192
#define B_  2
#define HW_ (H_*W_)

// ---------------------------------------------------------------------------
// Pack conv weights w[Cout][Cin][3][3] (f32) into per-lane WMMA A-fragment
// order (bf16): flat = ((octile*nch + ch)*32 + lane)*16 + i
// A 16x32 bf16 layout: lane hs=lane>>4, m=lane&15; elem i -> v=i>>1,j=i&1,
// k = 2v + j + 8*hs + (v>=4 ? 8 : 0)   (ISA 7.12.2, 16-bit A 16x32 table)
// ---------------------------------------------------------------------------
__global__ void pack_weights(const float* __restrict__ w, __bf16* __restrict__ apk,
                             int Cout, int Cin)
{
    const int K = Cin * 9;
    const int nch = K / 32;
    const int total = Cout * K;
    int t = blockIdx.x * blockDim.x + threadIdx.x;
    if (t >= total) return;
    int i    = t & 15;
    int lane = (t >> 4) & 31;
    int rest = t >> 9;              // octile*nch + ch
    int ch     = rest % nch;
    int octile = rest / nch;
    int hs = lane >> 4, m = lane & 15;
    int v = i >> 1, j = i & 1;
    int kl = 2 * v + j + 8 * hs + (v >= 4 ? 8 : 0);
    int kk = ch * 32 + kl;          // k = ic*9 + ky*3 + kx
    int ic = kk / 9, rem = kk - ic * 9;
    int ky = rem / 3, kx = rem - ky * 3;
    int oc = octile * 16 + m;
    apk[t] = (__bf16)w[((oc * Cin + ic) * 3 + ky) * 3 + kx];
}

__global__ void zero_stats(float* s)
{
    if (threadIdx.x < 384) s[threadIdx.x] = 0.f;
}

// ---------------------------------------------------------------------------
// 3x3 SAME conv as implicit GEMM over v_wmma_f32_16x16x32_bf16.
// Stage 1: halo (CIN x 3 x 18) -> LDS.
// Stage 2: block-cooperative im2col expansion into B-fragment order
//          (btile[ch][lane][16] bf16, 32B contiguous per lane) -> shared by
//          all oc tiles, loaded with ds_load_b128 in the K-loop.
// NW waves per block; waves < COUT/16 own one oc tile each.
// OMODE: 0 = relu->bf16, 1 = bf16, 2 = f32, 3 = f32 + addsrc (residual)
// ---------------------------------------------------------------------------
template<int CIN, int COUT, int NW, bool IN_BF16, int OMODE>
__global__ __launch_bounds__(NW*32)
void conv3x3_wmma(const void* __restrict__ inp, const __bf16* __restrict__ apk,
                  const float* __restrict__ bias, void* __restrict__ outp,
                  const float* __restrict__ addsrc)
{
    constexpr int K      = CIN * 9;
    constexpr int NCH    = K / 32;
    constexpr int NTILES = COUT / 16;

    __shared__ __bf16 halo [CIN * 3 * 20];  // [ic][ky(3)][col(18, pad 20)]
    __shared__ __bf16 btile[NCH * 512];     // [ch][lane(32)][16] fragment order

    const int tid   = threadIdx.x;
    const int strip = blockIdx.x;
    const int xt = strip % (W_ / 16);
    const int y  = (strip / (W_ / 16)) % H_;
    const int b  = strip / ((W_ / 16) * H_);
    const int x0 = xt * 16;

    // --- stage 1: halo load (rows y-1..y+1, cols x0-1..x0+16, zero pad) ---
    for (int t = tid; t < CIN * 54; t += NW * 32) {
        int ic = t / 54, rem = t - ic * 54;
        int r  = rem / 18, cc = rem - r * 18;
        int yy = y - 1 + r, xx = x0 - 1 + cc;
        float val = 0.f;
        if (yy >= 0 && yy < H_ && xx >= 0 && xx < W_) {
            size_t gi = ((size_t)(b * CIN + ic) * H_ + yy) * W_ + xx;
            val = IN_BF16 ? (float)((const __bf16*)inp)[gi]
                          : ((const float*)inp)[gi];
        }
        halo[ic * 60 + r * 20 + cc] = (__bf16)val;
    }
    __syncthreads();

    // --- stage 2: expand once into B-fragment order ---
    // B 32x16 bf16 layout: lane ln (hs=ln>>4, n=ln&15), elem i (v=i>>1,j=i&1)
    // -> k = 2v + j + 8*hs + (v>=4 ? 8 : 0);  value = in[ic][y+ky-1][x0+n+kx-1]
    for (int t = tid; t < NCH * 512; t += NW * 32) {
        int ch = t >> 9;
        int r  = t & 511;
        int ln = r >> 4, i = r & 15;
        int hs2 = ln >> 4, n2 = ln & 15;
        int v = i >> 1, j = i & 1;
        int kk = ch * 32 + 2 * v + j + 8 * hs2 + (v >= 4 ? 8 : 0);
        int ic = kk / 9; int rem = kk - ic * 9;
        int ky = rem / 3; int kx = rem - ky * 3;
        btile[t] = halo[ic * 60 + ky * 20 + n2 + kx];
    }
    __syncthreads();

    const int wave = tid >> 5;
    const int lane = tid & 31;

    if (wave < NTILES) {
        const int octile = wave;
        v8f acc = {};
        const __bf16* ap = apk + (size_t)(octile * NCH) * 512 + lane * 16;
        const __bf16* bp_ = btile + lane * 16;

        #pragma unroll 3
        for (int ch = 0; ch < NCH; ++ch) {
            v16bf afrag = *(const v16bf*)(ap  + (size_t)ch * 512);
            v16bf bfrag = *(const v16bf*)(bp_ + (size_t)ch * 512);
            acc = __builtin_amdgcn_wmma_f32_16x16x32_bf16(
                false, afrag, false, bfrag, (short)0, acc, false, false);
        }

        // D layout: VGPR r, lane -> M = r + 8*hs, N = lane&15
        const int hs = lane >> 4;
        const int n  = lane & 15;
        #pragma unroll
        for (int r = 0; r < 8; ++r) {
            int m  = r + 8 * hs;
            int oc = octile * 16 + m;
            size_t gi = ((size_t)(b * COUT + oc) * H_ + y) * W_ + x0 + n;
            float vo = acc[r] + bias[oc];
            if (OMODE == 0) { vo = vo > 0.f ? vo : 0.f; ((__bf16*)outp)[gi] = (__bf16)vo; }
            else if (OMODE == 1) { ((__bf16*)outp)[gi] = (__bf16)vo; }
            else if (OMODE == 2) { ((float*)outp)[gi] = vo; }
            else { ((float*)outp)[gi] = vo + addsrc[gi]; }
        }
    }
}

// ---------------------------------------------------------------------------
// Per-8x8-patch circular convolution of q (ch c) with k (ch 96+c) of hidden,
// equal to irfft2(fft2(q)*fft2(k)) for real inputs. One wave per patch.
// Also accumulates per-(b,c) sum / sumsq for the H*W LayerNorm.
// ---------------------------------------------------------------------------
__global__ __launch_bounds__(256)
void patch_circconv(const __bf16* __restrict__ hidden, float* __restrict__ cout,
                    float* __restrict__ ssum, float* __restrict__ ssq)
{
    __shared__ float qs[8][64];
    __shared__ float ks[8][64];
    const int wave = threadIdx.x >> 5;
    const int lane = threadIdx.x & 31;
    const int patch = blockIdx.x * 8 + wave;   // B_*96*24*24 total
    const int pw = patch % 24;
    const int ph = (patch / 24) % 24;
    const int c  = (patch / (24 * 24)) % 96;
    const int b  = patch / (24 * 24 * 96);

    const __bf16* qb = hidden + ((size_t)(b * 192 + c)      * H_ + ph * 8) * W_ + pw * 8;
    const __bf16* kb = hidden + ((size_t)(b * 192 + 96 + c) * H_ + ph * 8) * W_ + pw * 8;
    #pragma unroll
    for (int e = lane; e < 64; e += 32) {
        int r = e >> 3, cc = e & 7;
        qs[wave][e] = (float)qb[r * W_ + cc];
        ks[wave][e] = (float)kb[r * W_ + cc];
    }
    __syncthreads();

    float lsum = 0.f, lsq = 0.f;
    float* outb = cout + ((size_t)(b * 96 + c) * H_ + ph * 8) * W_ + pw * 8;
    #pragma unroll
    for (int e = lane; e < 64; e += 32) {
        int i = e >> 3, j = e & 7;
        float acc = 0.f;
        #pragma unroll
        for (int a = 0; a < 8; ++a) {
            int ia = ((i - a) & 7) * 8;
            #pragma unroll
            for (int bb = 0; bb < 8; ++bb)
                acc += qs[wave][a * 8 + bb] * ks[wave][ia + ((j - bb) & 7)];
        }
        outb[i * W_ + j] = acc;
        lsum += acc; lsq += acc * acc;
    }
    for (int m = 16; m >= 1; m >>= 1) {
        lsum += __shfl_xor(lsum, m, 32);
        lsq  += __shfl_xor(lsq,  m, 32);
    }
    if (lane == 0) {
        atomicAdd(&ssum[b * 96 + c], lsum);
        atomicAdd(&ssq [b * 96 + c], lsq);
    }
}

// amap = layernorm_{H,W}(cout) * fpan  (bf16 out, feeds conva WMMA)
__global__ void normalize_mul(const float* __restrict__ cout,
                              const float* __restrict__ fpan,
                              const float* __restrict__ ssum,
                              const float* __restrict__ ssq,
                              __bf16* __restrict__ amap, int total)
{
    int idx = blockIdx.x * blockDim.x + threadIdx.x;
    if (idx >= total) return;
    int plane = idx / HW_;                    // b*96 + c
    const float inv = 1.f / (float)HW_;
    float mu  = ssum[plane] * inv;
    float var = ssq[plane] * inv - mu * mu;
    float rs  = rsqrtf(var + 1e-5f);
    amap[idx] = (__bf16)((cout[idx] - mu) * rs * fpan[idx]);
}

// ---------------------------------------------------------------------------
extern "C" void kernel_launch(void* const* d_in, const int* in_sizes, int n_in,
                              void* d_out, int out_size, void* d_ws, size_t ws_size,
                              hipStream_t stream)
{
    const float* ms  = (const float*)d_in[0];
    const float* pan = (const float*)d_in[1];
    const float* w1  = (const float*)d_in[2];
    const float* b1  = (const float*)d_in[3];
    const float* w2  = (const float*)d_in[4];
    const float* b2  = (const float*)d_in[5];
    const float* wp  = (const float*)d_in[6];
    const float* bp  = (const float*)d_in[7];
    const float* wa  = (const float*)d_in[8];
    const float* ba  = (const float*)d_in[9];

    char* ws = (char*)d_ws;
    size_t off = 0;
    auto take = [&](size_t bytes) {
        char* p = ws + off;
        off = (off + bytes + 255) & ~(size_t)255;
        return p;
    };
    __bf16* apk1   = (__bf16*)take((size_t)96  * 288 * 2);
    __bf16* apk2   = (__bf16*)take((size_t)192 * 864 * 2);
    __bf16* apkp   = (__bf16*)take((size_t)96  * 288 * 2);
    __bf16* apka   = (__bf16*)take((size_t)32  * 864 * 2);
    float*  stats  = (float*) take(384 * 4);           // ssum[192], ssq[192]
    __bf16* t1     = (__bf16*)take((size_t)B_ * 96  * HW_ * 2);
    __bf16* hidden = (__bf16*)take((size_t)B_ * 192 * HW_ * 2);
    float*  fpan   = (float*) take((size_t)B_ * 96  * HW_ * 4);
    float*  cout_  = (float*) take((size_t)B_ * 96  * HW_ * 4);
    __bf16* amap   = (__bf16*)take((size_t)B_ * 96  * HW_ * 2);

    // weight packing + stat reset (deterministic each call; ws is volatile)
    pack_weights<<<(96  * 288 + 255) / 256, 256, 0, stream>>>(w1, apk1, 96, 32);
    pack_weights<<<(192 * 864 + 255) / 256, 256, 0, stream>>>(w2, apk2, 192, 96);
    pack_weights<<<(96  * 288 + 255) / 256, 256, 0, stream>>>(wp, apkp, 96, 32);
    pack_weights<<<(32  * 864 + 255) / 256, 256, 0, stream>>>(wa, apka, 32, 96);
    zero_stats<<<1, 384, 0, stream>>>(stats);

    const int strips = B_ * H_ * (W_ / 16);            // 4608

    // t1 = relu(conv(ms, w1) + b1)           32 -> 96, bf16
    conv3x3_wmma<32, 96, 6, false, 0><<<strips, 192, 0, stream>>>(ms, apk1, b1, t1, nullptr);
    // hidden = conv(t1, w2) + b2             96 -> 192, bf16 (q | k)
    conv3x3_wmma<96, 192, 12, true, 1><<<strips, 384, 0, stream>>>(t1, apk2, b2, hidden, nullptr);
    // fpan = conv(pan, wp) + bp              32 -> 96, f32
    conv3x3_wmma<32, 96, 6, false, 2><<<strips, 192, 0, stream>>>(pan, apkp, bp, fpan, nullptr);

    // per-patch circular conv (== FFT stage) + LN statistics
    const int patches = B_ * 96 * (H_ / 8) * (W_ / 8); // 110592
    patch_circconv<<<patches / 8, 256, 0, stream>>>(hidden, cout_, stats, stats + 192);

    // amap = layernorm(cout) * fpan (bf16)
    const int tot = B_ * 96 * HW_;
    normalize_mul<<<(tot + 255) / 256, 256, 0, stream>>>(cout_, fpan, stats, stats + 192,
                                                         amap, tot);

    // out = conv(amap, wa) + ba + ms         96 -> 32, f32 into d_out
    conv3x3_wmma<96, 32, 8, true, 3><<<strips, 256, 0, stream>>>(amap, apka, ba,
                                                                 d_out, ms);
}
